// Multi_Head_Attention_with_RoPE_26895085208062
// MI455X (gfx1250) — matrix operand order repaired and checked by reading, not compiled, not run on hardware
//
#include <hip/hip_runtime.h>

// ---- fixed problem dims ----
#define B_ 2
#define T_ 2048
#define C_ 1024
#define H_ 16
#define D_ 64

typedef __attribute__((ext_vector_type(16))) __bf16 v16bf;
typedef __attribute__((ext_vector_type(8)))  float  v8f;
typedef __attribute__((ext_vector_type(4)))  int    v4i;

// float -> bf16 bits, round-to-nearest-even
__device__ __forceinline__ unsigned short f2bfu(float f) {
    union { float f; unsigned u; } a; a.f = f;
    unsigned r = a.u + 0x7FFFu + ((a.u >> 16) & 1u);
    return (unsigned short)(r >> 16);
}
__device__ __forceinline__ float bf2f(unsigned short s) {
    union { unsigned u; float f; } x; x.u = ((unsigned)s) << 16;
    return x.f;
}

__device__ __forceinline__ v8f v8f_zero() {
    v8f z;
#pragma unroll
    for (int r = 0; r < 8; ++r) z[r] = 0.0f;
    return z;
}

// A-operand (16x32 bf16, M x K): lane holds row m=lane%16;
// elements 0..7  = K: kb + (lane/16)*8 .. +7
// elements 8..15 = K: kb + 16 + (lane/16)*8 .. +7
// p points at &row[kb + (lane/16)*8]
__device__ __forceinline__ v16bf make_a(const unsigned short* p) {
    union { v16bf v; v4i q[2]; } u;
    u.q[0] = *(const v4i*)(p);
    u.q[1] = *(const v4i*)(p + 16);
    return u.v;
}

// B-operand (32x16 bf16, K x N): lane holds column n=lane%16;
// elements 0..7  = K: kb + (lane/16)*8 .. +7
// elements 8..15 = K: kb + 16 + (lane/16)*8 .. +7
// p points at &col_as_row[kb + (lane/16)*8]  (B stored transposed: N rows, K cols)
__device__ __forceinline__ v16bf make_b(const unsigned short* p) {
    union { v16bf v; v4i q[2]; } u;
    u.q[0] = *(const v4i*)(p);
    u.q[1] = *(const v4i*)(p + 16);
    return u.v;
}

__device__ __forceinline__ v8f wmma_bf16(v16bf a, v16bf b, v8f c) {
    return __builtin_amdgcn_wmma_f32_16x16x32_bf16(false, a, false, b, (short)0, c,
                                                   false, false);
}

// ---------------- kernel 1: fp32 -> bf16 ----------------
__global__ void f32_to_bf16_k(const float* __restrict__ in,
                              unsigned short* __restrict__ out, int n) {
    int i = blockIdx.x * blockDim.x + threadIdx.x;
    if (i < n) out[i] = f2bfu(in[i]);
}

// ---------------- kernel 2/5: C = A(MxK) * B(NxK)^T ----------------
// A, B bf16 row-major (K contiguous). Wave computes 32x64 tile; 4 waves stack M.
template <bool OUT_F32>
__global__ __launch_bounds__(128) void gemm_bf16_nt(
    const unsigned short* __restrict__ A, const unsigned short* __restrict__ Bm,
    void* __restrict__ Out, int M, int N, int K) {
    const int lane = threadIdx.x & 31;
    const int wave = threadIdx.x >> 5;
    const int l16  = lane & 15;
    const int hh   = lane >> 4;
    const int h8   = hh * 8;
    const int h16  = hh * 16;
    const int mb   = blockIdx.y * 128 + wave * 32;
    const int nb   = blockIdx.x * 64;

    v8f acc[2][4];
#pragma unroll
    for (int i = 0; i < 2; ++i)
#pragma unroll
        for (int j = 0; j < 4; ++j) acc[i][j] = v8f_zero();

    for (int kb = 0; kb < K; kb += 32) {
        v16bf a[2], b[4];
#pragma unroll
        for (int i = 0; i < 2; ++i)
            a[i] = make_a(A + (size_t)(mb + i * 16 + l16) * K + kb + h8);
#pragma unroll
        for (int j = 0; j < 4; ++j)
            b[j] = make_b(Bm + (size_t)(nb + j * 16 + l16) * K + kb + h8);
#pragma unroll
        for (int i = 0; i < 2; ++i)
#pragma unroll
            for (int j = 0; j < 4; ++j) acc[i][j] = wmma_bf16(a[i], b[j], acc[i][j]);
    }

#pragma unroll
    for (int i = 0; i < 2; ++i)
#pragma unroll
        for (int j = 0; j < 4; ++j)
#pragma unroll
            for (int r = 0; r < 8; ++r) {
                int m = mb + i * 16 + r + 8 * hh;
                int n = nb + j * 16 + l16;
                float v = acc[i][j][r];
                if (OUT_F32)
                    ((float*)Out)[(size_t)m * N + n] = v;
                else
                    ((unsigned short*)Out)[(size_t)m * N + n] = f2bfu(v);
            }
}

// ---------------- kernel 3: RoPE + head repack ----------------
// qkv: (B*T, 3C) bf16.  Q,K out: (B,H,T,D) bf16 with RoPE.  Vt out: (B,H,D,T) bf16.
__global__ void rope_pack(const unsigned short* __restrict__ qkv,
                          unsigned short* __restrict__ Q,
                          unsigned short* __restrict__ Kk,
                          unsigned short* __restrict__ Vt) {
    int i = blockIdx.x * blockDim.x + threadIdx.x;  // < B*T*H*(D/2)
    int j = i & 31;              // pair index within head dim
    int h = (i >> 5) & (H_ - 1);
    int t = (i >> 9) & (T_ - 1);
    int b = i >> 20;

    size_t row = (size_t)(b * T_ + t) * (3 * C_);
    int col = h * D_ + 2 * j;
    float qr = bf2f(qkv[row + col]);
    float qi = bf2f(qkv[row + col + 1]);
    float kr = bf2f(qkv[row + C_ + col]);
    float ki = bf2f(qkv[row + C_ + col + 1]);
    float vr = bf2f(qkv[row + 2 * C_ + col]);
    float vi = bf2f(qkv[row + 2 * C_ + col + 1]);

    // inv_freq = theta^(-2j/D) = exp(-j * ln(theta)/32)
    float invf = __expf(-(float)j * (9.210340371976184f / 32.0f));
    float ang = (float)t * invf;
    float s, c;
    sincosf(ang, &s, &c);  // accurate range reduction (ang up to ~2047 rad)

    size_t bh = (size_t)(b * H_ + h);
    size_t qo = (bh * T_ + t) * D_ + 2 * j;
    Q[qo]      = f2bfu(qr * c - qi * s);
    Q[qo + 1]  = f2bfu(qr * s + qi * c);
    Kk[qo]     = f2bfu(kr * c - ki * s);
    Kk[qo + 1] = f2bfu(kr * s + ki * c);

    size_t vo = (bh * D_ + 2 * j) * T_ + t;
    Vt[vo]      = f2bfu(vr);
    Vt[vo + T_] = f2bfu(vi);
}

// ---------------- kernel 4: causal flash attention ----------------
// One wave (32 threads) per 16-row q tile. K-tiles of 32 columns.
__global__ __launch_bounds__(32) void attn_kernel(
    const unsigned short* __restrict__ Q, const unsigned short* __restrict__ Kk,
    const unsigned short* __restrict__ Vt, unsigned short* __restrict__ Ob) {
    __shared__ unsigned short pLds[16 * 40];  // 16 rows, stride 40 (16B-aligned rows)

    const int lane = threadIdx.x & 31;
    const int l16  = lane & 15;
    const int hh   = lane >> 4;
    const int h8   = hh * 8;
    const int h16  = hh * 16;
    const int bh   = blockIdx.y;
    const int b    = bh >> 4;
    const int h    = bh & 15;
    const int q0   = blockIdx.x * 16;

    const unsigned short* Qp = Q + ((size_t)bh * T_ + q0) * D_;
    const unsigned short* Kp = Kk + (size_t)bh * T_ * D_;
    const unsigned short* Vp = Vt + (size_t)bh * D_ * T_;

    // Q tile as A operands: K-dim D=64 -> two chunks of 32
    v16bf aq0 = make_a(Qp + (size_t)l16 * D_ + h8);
    v16bf aq1 = make_a(Qp + (size_t)l16 * D_ + 32 + h8);

    float mrun[8], lrun[8], alpha[8];
#pragma unroll
    for (int r = 0; r < 8; ++r) { mrun[r] = -1e30f; lrun[r] = 0.0f; }
    v8f accO[4];
#pragma unroll
    for (int n = 0; n < 4; ++n) accO[n] = v8f_zero();

    const float scale = 0.125f;  // 1/sqrt(64)
    const int kEnd = q0 + 16;

    for (int kb = 0; kb < kEnd; kb += 32) {
        // S = scale * Q K^T over 32 key columns (two 16-wide WMMA n-tiles)
        v8f S[2];
#pragma unroll
        for (int jt = 0; jt < 2; ++jt) {
            const unsigned short* Kr = Kp + (size_t)(kb + jt * 16 + l16) * D_;
            v16bf b0 = make_b(Kr + h8);        // K-rows are B^T rows -> B layout
            v16bf b1 = make_b(Kr + 32 + h8);
            v8f c = v8f_zero();
            c = wmma_bf16(aq0, b0, c);
            c = wmma_bf16(aq1, b1, c);
#pragma unroll
            for (int r = 0; r < 8; ++r) {
                int mg = q0 + r + 8 * hh;
                int ng = kb + jt * 16 + l16;
                float sv = c[r] * scale;
                if (ng > mg) sv = -1e30f;  // causal mask
                c[r] = sv;
            }
            S[jt] = c;
        }

        // online softmax per row (row = r + 8*hh, 16 lanes per row-half)
#pragma unroll
        for (int r = 0; r < 8; ++r) {
            float mx = fmaxf(S[0][r], S[1][r]);
            mx = fmaxf(mx, __shfl_xor(mx, 1, 32));
            mx = fmaxf(mx, __shfl_xor(mx, 2, 32));
            mx = fmaxf(mx, __shfl_xor(mx, 4, 32));
            mx = fmaxf(mx, __shfl_xor(mx, 8, 32));
            float mnew = fmaxf(mrun[r], mx);
            alpha[r] = __expf(mrun[r] - mnew);
            mrun[r] = mnew;
            float p0 = __expf(S[0][r] - mnew);
            float p1 = __expf(S[1][r] - mnew);
            float rs = p0 + p1;
            rs += __shfl_xor(rs, 1, 32);
            rs += __shfl_xor(rs, 2, 32);
            rs += __shfl_xor(rs, 4, 32);
            rs += __shfl_xor(rs, 8, 32);
            lrun[r] = lrun[r] * alpha[r] + rs;
            int mrow = r + 8 * hh;
            pLds[mrow * 40 + l16]      = f2bfu(p0);   // cols 0..15 of this k-tile
            pLds[mrow * 40 + 16 + l16] = f2bfu(p1);   // cols 16..31
        }
#pragma unroll
        for (int n = 0; n < 4; ++n)
#pragma unroll
            for (int r = 0; r < 8; ++r) accO[n][r] *= alpha[r];

        __syncthreads();
        // reload P in A-operand layout (row = l16, K-chunks h8 / 16+h8)
        union { v16bf v; unsigned short s[16]; } up;
        {
            const int row = l16 * 40;
#pragma unroll
            for (int e = 0; e < 8; ++e) up.s[e] = pLds[row + h8 + e];
#pragma unroll
            for (int e = 0; e < 8; ++e) up.s[8 + e] = pLds[row + 16 + h8 + e];
        }
        __syncthreads();

        // O += P * V  (B operand from transposed V: row d contiguous in t)
#pragma unroll
        for (int n = 0; n < 4; ++n) {
            const unsigned short* Vr = Vp + (size_t)(n * 16 + l16) * T_ + kb;
            v16bf bv = make_b(Vr + h8);
            accO[n] = wmma_bf16(up.v, bv, accO[n]);
        }
    }

    // write O / l as bf16 into (B, T, H*D) = (B*T, C) layout
#pragma unroll
    for (int n = 0; n < 4; ++n)
#pragma unroll
        for (int r = 0; r < 8; ++r) {
            int mrow = r + 8 * hh;
            float o = accO[n][r] / lrun[r];
            size_t off = (size_t)(b * T_ + q0 + mrow) * C_ + h * D_ + n * 16 + l16;
            Ob[off] = f2bfu(o);
        }
}

// ---------------- launcher ----------------
extern "C" void kernel_launch(void* const* d_in, const int* in_sizes, int n_in,
                              void* d_out, int out_size, void* d_ws, size_t ws_size,
                              hipStream_t stream) {
    (void)in_sizes; (void)n_in; (void)out_size; (void)ws_size;
    const float* x     = (const float*)d_in[0];
    const float* w_qkv = (const float*)d_in[1];
    const float* w_out = (const float*)d_in[2];

    char* ws = (char*)d_ws;
    unsigned short* xb    = (unsigned short*)(ws);                        //  8 MB
    unsigned short* wqkvb = (unsigned short*)(ws + (size_t)8  * 1048576); //  6 MB
    unsigned short* woutb = (unsigned short*)(ws + (size_t)14 * 1048576); //  2 MB
    unsigned short* qkvb  = (unsigned short*)(ws + (size_t)16 * 1048576); // 24 MB
    unsigned short* Qb    = (unsigned short*)(ws + (size_t)40 * 1048576); //  8 MB
    unsigned short* Kb    = (unsigned short*)(ws + (size_t)48 * 1048576); //  8 MB
    unsigned short* Vt    = (unsigned short*)(ws + (size_t)56 * 1048576); //  8 MB
    unsigned short* attnb = xb;  // x dead after QKV GEMM; reuse its slot

    const int M = B_ * T_;  // 4096

    int nx = M * C_;
    f32_to_bf16_k<<<(nx + 255) / 256, 256, 0, stream>>>(x, xb, nx);
    int nw = 3 * C_ * C_;
    f32_to_bf16_k<<<(nw + 255) / 256, 256, 0, stream>>>(w_qkv, wqkvb, nw);
    int no = C_ * C_;
    f32_to_bf16_k<<<(no + 255) / 256, 256, 0, stream>>>(w_out, woutb, no);

    gemm_bf16_nt<false><<<dim3(3 * C_ / 64, M / 128), 128, 0, stream>>>(
        xb, wqkvb, qkvb, M, 3 * C_, C_);

    int nr = B_ * T_ * H_ * (D_ / 2);
    rope_pack<<<nr / 256, 256, 0, stream>>>(qkvb, Qb, Kb, Vt);

    attn_kernel<<<dim3(T_ / 16, B_ * H_), 32, 0, stream>>>(Qb, Kb, Vt, attnb);

    gemm_bf16_nt<true><<<dim3(C_ / 64, M / 128), 128, 0, stream>>>(
        attnb, woutb, d_out, M, C_, C_);
}
